// Model_65953517797504
// MI455X (gfx1250) — matrix operand order repaired and checked by reading, not compiled, not run on hardware
//
#include <hip/hip_runtime.h>
#include <hip/hip_bf16.h>
#include <stdint.h>

#define LABEL    64
#define SEQ      512
#define BATCHSZ  512
#define START_ID 62
#define PAD_ID   63
#define LN2F     0.69314718055994531f

typedef __attribute__((ext_vector_type(16))) _Float16 v16h;
typedef __attribute__((ext_vector_type(8)))  _Float16 v8h;
typedef __attribute__((ext_vector_type(8)))  float    v8f;
typedef __attribute__((ext_vector_type(4)))  unsigned int u32x4;
typedef __attribute__((ext_vector_type(8)))  int      i32x8;
typedef __attribute__((ext_vector_type(4)))  int      i32x4;

__device__ __forceinline__ unsigned lds_off_of(const void* p) {
  // LDS flat address: low 32 bits are the LDS byte offset (aperture in high bits)
  return (unsigned)(unsigned long long)(uintptr_t)p;
}

// TDM: async 2-D tile load (16 rows x 64 f32, row stride SEQ*LABEL elems) -> LDS.
__device__ __forceinline__ void tdm_load_tile(const float* gsrc, unsigned lds_off) {
  unsigned long long ga = (unsigned long long)(uintptr_t)gsrc;
  u32x4 g0;
  g0[0] = 1u;                                              // count=1, user desc
  g0[1] = lds_off;                                         // lds_addr
  g0[2] = (unsigned)(ga & 0xffffffffull);                  // global_addr[31:0]
  g0[3] = (unsigned)((ga >> 32) & 0x1ffffffull) | (2u << 30); // addr[56:32] | type=2
  i32x8 g1;
  g1[0] = (int)(2u << 16);            // workgroup_mask=0, data_size=2 (4B)
  g1[1] = (int)(64u << 16);           // tensor_dim0 = 64 (bits 79:48, low half)
  g1[2] = (int)(16u << 16);           // tensor_dim0 hi=0 | tensor_dim1 = 16
  g1[3] = (int)(64u << 16);           // tensor_dim1 hi=0 | tile_dim0 = 64
  g1[4] = (int)16;                    // tile_dim1 = 16, tile_dim2 = 0
  g1[5] = (int)(SEQ * LABEL);         // tensor_dim0_stride = 32768 elems
  g1[6] = 0;
  g1[7] = 0;
  i32x4 gz4 = {0, 0, 0, 0};           // 2-D tensor: groups 2/3 unused
  i32x8 gz8 = {0, 0, 0, 0, 0, 0, 0, 0};
  __builtin_amdgcn_tensor_load_to_lds(g0, g1, gz4, gz4, gz8, 0);
}

// One wave (32 lanes) per 16-batch tile. Alpha kept in WMMA C/D layout:
// acc[t][r] : lane l holds alpha[batch = (l<16 ? r : r+8)][label = t*16 + l%16].
__global__ __launch_bounds__(32) void crf_forward_kernel(
    const float* __restrict__ ts, const float* __restrict__ T,
    const int* __restrict__ lengths, float* __restrict__ fwd_partial) {
  __shared__ __align__(16) _Float16 Elds[LABEL][LABEL];   // Elds[j][i] = exp(T[i][j])
  __shared__ __align__(16) _Float16 Albs[16][LABEL];      // A transpose staging
  __shared__ __align__(16) float    ebuf[2][16][LABEL];   // emission double buffer

  const int lane = threadIdx.x;
  const int half = lane >> 4;
  const int ln   = lane & 15;
  const int b0   = blockIdx.x * 16;

  // Build exp(T) column-major in LDS (so B-fragment K pairs are contiguous).
  for (int idx = lane; idx < LABEL * LABEL; idx += 32) {
    int i = idx >> 6, j = idx & 63;
    Elds[j][i] = (_Float16)__expf(T[i * LABEL + j]);
  }
  __syncthreads();

  // Hoist B fragments (exp(T), 64x64 f16) into registers for all 512 steps.
  // B k order used here (contiguous, as Af): VGPR v, lanes 0-15: K=2v,2v+1 ; lanes 16-31: K=16+2v,17+2v.
  v16h Bf[2][4];
  for (int c = 0; c < 2; ++c)
    for (int t = 0; t < 4; ++t) {
      int n = t * 16 + ln;
      v16h bv;
      for (int v = 0; v < 8; ++v) {
        int k0 = c * 32 + half * 16 + 2 * v;
        bv[2 * v]     = Elds[n][k0];
        bv[2 * v + 1] = Elds[n][k0 + 1];
      }
      Bf[c][t] = bv;
    }

  int lenr[8];
  for (int r = 0; r < 8; ++r) lenr[r] = lengths[b0 + half * 8 + r];
  int maxLen = lengths[b0 + ln];
  for (int d = 1; d < 32; d <<= 1) {
    int o = __shfl_xor(maxLen, d, 32);
    maxLen = maxLen > o ? maxLen : o;
  }

  // alpha0 = tag_scores[:,0,:] + T[START,:]
  v8f acc[4];
  for (int t = 0; t < 4; ++t) {
    int j = t * 16 + ln;
    float tj = T[START_ID * LABEL + j];
    for (int r = 0; r < 8; ++r) {
      int m = half * 8 + r;
      acc[t][r] = ts[((size_t)(b0 + m) * SEQ) * LABEL + j] + tj;
    }
  }

  const unsigned eoff[2] = { lds_off_of(&ebuf[0][0][0]), lds_off_of(&ebuf[1][0][0]) };
  if (maxLen > 1)
    tdm_load_tile(ts + ((size_t)b0 * SEQ + 1) * LABEL, eoff[0]);

  for (int s = 1; s < maxLen; ++s) {
    if (s + 1 < maxLen) {
      tdm_load_tile(ts + ((size_t)b0 * SEQ + (s + 1)) * LABEL, eoff[s & 1]);
      __builtin_amdgcn_s_wait_tensorcnt(1);   // older (step s) tile complete
    } else {
      __builtin_amdgcn_s_wait_tensorcnt(0);
    }
    asm volatile("" ::: "memory");
    const float (*eb)[LABEL] = ebuf[(s - 1) & 1];

    // per-batch max over labels (4 tiles + xor-shuffle within 16-lane halves)
    float mr[8];
    for (int r = 0; r < 8; ++r) {
      float v = fmaxf(fmaxf(acc[0][r], acc[1][r]), fmaxf(acc[2][r], acc[3][r]));
      for (int d = 1; d < 16; d <<= 1) v = fmaxf(v, __shfl_xor(v, d, 32));
      mr[r] = v;
    }

    // A = exp(alpha - m) -> LDS (transpose from D layout to A layout)
    for (int t = 0; t < 4; ++t) {
      int j = t * 16 + ln;
      for (int r = 0; r < 8; ++r)
        Albs[half * 8 + r][j] = (_Float16)__expf(acc[t][r] - mr[r]);
    }
    __syncthreads();  // single-wave WG: lowers to NOP; same-wave LDS is in-order

    // A fragments: lanes<16 take K = c*32+{0..15}, lanes>=16 take K = c*32+{16..31}.
    v16h Af[2];
    for (int c = 0; c < 2; ++c) {
      int kA = c * 32 + half * 16;
      v8h lo = *(const v8h*)&Albs[ln][kA];
      v8h hi = *(const v8h*)&Albs[ln][kA + 8];
      Af[c] = __builtin_shufflevector(lo, hi, 0,1,2,3,4,5,6,7,8,9,10,11,12,13,14,15);
    }

    // q = A x exp(T) (K=64 via two chained WMMAs), then alpha' = e + m + log q.
    // q in [1e-30, ~1e2]: never subnormal/inf, so raw v_log_f32 (log2) + one
    // fma by ln2 replaces the guarded __logf expansion.
    for (int t = 0; t < 4; ++t) {
      v8f q = {};
      q = __builtin_amdgcn_wmma_f32_16x16x32_f16(false, Af[0], false, Bf[0][t],
                                                 (short)0, q, false, false);
      q = __builtin_amdgcn_wmma_f32_16x16x32_f16(false, Af[1], false, Bf[1][t],
                                                 (short)0, q, false, false);
      int j = t * 16 + ln;
      for (int r = 0; r < 8; ++r) {
        float l2 = __log2f(fmaxf(q[r], 1e-30f));
        float na = __builtin_fmaf(LN2F, l2, eb[half * 8 + r][j] + mr[r]);
        // Pin 'na' so the ternary lowers to a single v_cndmask instead of
        // per-element exec-mask branches around the log chain.
        asm("" : "+v"(na));
        acc[t][r] = (s < lenr[r]) ? na : acc[t][r];
      }
    }
  }

  // final[b] = logsumexp_j(alpha[b,j] + T[j,PAD])
  float tp[4];
  for (int t = 0; t < 4; ++t) tp[t] = T[(t * 16 + ln) * LABEL + PAD_ID];
  float part = 0.f;
  for (int r = 0; r < 8; ++r) {
    float v0 = acc[0][r] + tp[0], v1 = acc[1][r] + tp[1];
    float v2 = acc[2][r] + tp[2], v3 = acc[3][r] + tp[3];
    float m2 = fmaxf(fmaxf(v0, v1), fmaxf(v2, v3));
    for (int d = 1; d < 16; d <<= 1) m2 = fmaxf(m2, __shfl_xor(m2, d, 32));
    float ss = __expf(v0 - m2) + __expf(v1 - m2) + __expf(v2 - m2) + __expf(v3 - m2);
    for (int d = 1; d < 16; d <<= 1) ss += __shfl_xor(ss, d, 32);
    part += __builtin_fmaf(LN2F, __log2f(ss), m2);
  }
  part += __shfl_xor(part, 16, 32);  // combine halves (batches 0-7 + 8-15)
  if (lane == 0) fwd_partial[blockIdx.x] = part;
}

// Gold path score: one block per batch, deterministic in-block reduction.
__global__ __launch_bounds__(64) void crf_gold_kernel(
    const float* __restrict__ ts, const float* __restrict__ T,
    const int* __restrict__ labels, const int* __restrict__ lengths,
    float* __restrict__ gold_partial) {
  int b = blockIdx.x;
  int len = lengths[b];
  float acc = 0.f;
  for (int s = threadIdx.x; s < len; s += 64) {
    int lab  = labels[(size_t)b * SEQ + s];
    int prev = (s == 0) ? START_ID : labels[(size_t)b * SEQ + s - 1];
    acc += T[prev * LABEL + lab] + ts[((size_t)b * SEQ + s) * LABEL + lab];
  }
  __shared__ float red[64];
  red[threadIdx.x] = acc;
  __syncthreads();
  if (threadIdx.x == 0) {
    float t = 0.f;
    for (int i = 0; i < 64; ++i) t += red[i];
    int last = labels[(size_t)b * SEQ + len - 1];
    t += T[last * LABEL + PAD_ID];
    gold_partial[b] = t;
  }
}

__global__ void crf_final_kernel(const float* __restrict__ fwd_partial,
                                 const float* __restrict__ gold_partial,
                                 float* __restrict__ out) {
  if (threadIdx.x == 0 && blockIdx.x == 0) {
    float f = 0.f;
    for (int i = 0; i < BATCHSZ / 16; ++i) f += fwd_partial[i];
    float g = 0.f;
    for (int i = 0; i < BATCHSZ; ++i) g += gold_partial[i];
    out[0] = (f - g) / (float)BATCHSZ;
  }
}

extern "C" void kernel_launch(void* const* d_in, const int* in_sizes, int n_in,
                              void* d_out, int out_size, void* d_ws, size_t ws_size,
                              hipStream_t stream) {
  const float* ts      = (const float*)d_in[0];   // [512,512,64]
  const float* T       = (const float*)d_in[1];   // [64,64]
  const int*   labels  = (const int*)d_in[2];     // [512,512]
  const int*   lengths = (const int*)d_in[3];     // [512]
  float* out = (float*)d_out;

  float* fwd_partial  = (float*)d_ws;             // 32 floats
  float* gold_partial = fwd_partial + 32;         // 512 floats

  crf_forward_kernel<<<BATCHSZ / 16, 32, 0, stream>>>(ts, T, lengths, fwd_partial);
  crf_gold_kernel<<<BATCHSZ, 64, 0, stream>>>(ts, T, labels, lengths, gold_partial);
  crf_final_kernel<<<1, 1, 0, stream>>>(fwd_partial, gold_partial, out);
  (void)in_sizes; (void)n_in; (void)out_size; (void)ws_size;
}
